// SimpleBlock_67997922230598
// MI455X (gfx1250) — matrix operand order repaired and checked by reading, not compiled, not run on hardware
//
#include <hip/hip_runtime.h>
#include <hip/hip_bf16.h>

// ---------------- types ----------------
typedef __attribute__((ext_vector_type(16))) __bf16        v16bf;
typedef __attribute__((ext_vector_type(8)))  float         v8f;
typedef __attribute__((ext_vector_type(4)))  unsigned int  v4u;

struct S32 { v4u lo, hi; };   // 32 bytes, trivially copyable, for bit_cast to v16bf

// ---------------- problem constants ----------------
#define N_PTS   150000
#define NNB     16
#define KPTS    15
#define CIN     64
#define COUT    64
#define NTILES  (N_PTS / 16)            // 9375 tiles of 16 points
#define WPG     4                       // waves per workgroup
#define NKB     32                      // padded contraction 1024 / 32
#define FK_STRIDE_US 1032               // ushorts per FK row (1024 + 8 pad -> 2064B, bank-friendly)
#define WL_BYTES 131072                 // swizzled bf16 W: 32 kb * 4 nt * 32 lanes * 32B
#define PW_BYTES 35328                  // per-wave: FK 33024 + FXS 2048 + REL 192 (+pad)
#define SMEM_BYTES (WL_BYTES + WPG * PW_BYTES)   // 272384 B < 320KB WGP LDS

static __device__ __forceinline__ unsigned short f2bfbits(float f) {
    unsigned u = __builtin_bit_cast(unsigned, f);
    unsigned r = u + 0x7FFFu + ((u >> 16) & 1u);       // round-to-nearest-even
    return (unsigned short)(r >> 16);
}

// Wave-local LDS fence: REL/FXS/FK are wave-private, so a full workgroup
// s_barrier is wasteful. DS ops from one wave issue in order on CDNA5; we only
// need to stop the compiler reordering LDS accesses across this point and wait
// for outstanding DS ops (split-counter s_wait_dscnt).
static __device__ __forceinline__ void wave_lds_sync() {
    __builtin_amdgcn_wave_barrier();
    asm volatile("s_wait_dscnt 0x0" ::: "memory");
    __builtin_amdgcn_wave_barrier();
}

// =====================================================================
// Main kernel: per 16-point tile, two WMMA GEMMs; writes pre-BN output.
// =====================================================================
__global__ __launch_bounds__(128)
void kpconv_main(const float* __restrict__ pos,
                 const float* __restrict__ x,
                 const int*   __restrict__ eidx,
                 const float* __restrict__ kp,
                 const float* __restrict__ Wg,
                 float*       __restrict__ out)
{
    extern __shared__ char smem[];
    unsigned short* Wl = (unsigned short*)smem;

    const int tid  = threadIdx.x;
    const int lane = tid & 31;
    const int warp = tid >> 5;
    const int lo16 = lane & 15;
    const int hi   = lane >> 4;         // lane half: 0 or 1

    // ---- cooperative load of W -> LDS, bf16, pre-swizzled into WMMA B-fragments ----
    // B fragment in this file's contiguous k order (same as A in phase C): lane = col%16 + 16*(K-half); element j <-> K offset j.
    for (int i = tid; i < KPTS * CIN * COUT; i += 128) {       // coalesced source reads
        int K  = i >> 6;                // 0..959  (= k*64 + c)
        int o  = i & 63;                // output channel
        int kb = K >> 5;
        int r  = K & 31;
        int lh = r >> 4;
        int j  = r & 15;
        int nt = o >> 4;
        int ln = lh * 16 + (o & 15);
        Wl[(((kb * 4 + nt) * 32) + ln) * 16 + j] = f2bfbits(Wg[i]);
    }
    for (int i = tid; i < 4096; i += 128) Wl[61440 + i] = 0;   // zero K = 960..1023 pad
    __syncthreads();   // cross-wave: W must be complete before any wave reads it

    char* wbase = smem + WL_BYTES + warp * PW_BYTES;
    unsigned short* FK  = (unsigned short*)wbase;               // [16][FK_STRIDE_US] bf16
    unsigned short* FXS = (unsigned short*)(wbase + 33024);     // [64 ch][16 nbr] bf16
    float*          REL = (float*)(wbase + 33024 + 2048);       // [16 nbr][3]

    const int  tile   = blockIdx.x * WPG + warp;
    const bool active = tile < NTILES;

    // per-lane kernel point (A-matrix row index = lane%16; row 15 is the zero pad)
    float kx = 1e9f, ky = 1e9f, kz = 1e9f;
    if (lo16 < KPTS) { kx = kp[lo16*3+0]; ky = kp[lo16*3+1]; kz = kp[lo16*3+2]; }

    const S32 z32 = {};
    const v8f zc  = __builtin_bit_cast(v8f, z32);

    // ---------------- Phase A+B: weights + fk = w^T @ fx per point ----------------
    for (int p = 0; p < 16; ++p) {
        const int n = tile * 16 + p;

        int id = N_PTS;                                       // shadow neighbor default
        if (active && lane < 16) id = eidx[n * NNB + lane];

        if (lane < 16) {
            float cx = 0.f, cy = 0.f, cz = 0.f;
            if (active) { cx = pos[n*4+1]; cy = pos[n*4+2]; cz = pos[n*4+3]; }
            float nx = 1e6f, ny = 1e6f, nz = 1e6f;            // SHADOW_COORD
            if (active && id < N_PTS) {
                const float4 pv = *(const float4*)(pos + (size_t)id * 4);
                nx = pv.y; ny = pv.z; nz = pv.w;
            }
            REL[lane*3+0] = nx - cx;
            REL[lane*3+1] = ny - cy;
            REL[lane*3+2] = nz - cz;
        }

        // gather neighbor features, convert bf16, store transposed: FXS[c][e]
        #pragma unroll 4
        for (int e = 0; e < 16; ++e) {
            const int ide = __shfl(id, e, 32);                // wave-uniform row index
            float vx = 0.f, vy = 0.f;
            if (active && ide < N_PTS) {
                const float2 t2 = *(const float2*)(x + (size_t)ide * CIN + lane * 2);
                vx = t2.x; vy = t2.y;
            }
            FXS[(lane*2 + 0) * 16 + e] = f2bfbits(vx);
            FXS[(lane*2 + 1) * 16 + e] = f2bfbits(vy);
        }
        wave_lds_sync();   // REL/FXS stores (this wave) -> reads (this wave)

        // A fragment: w[e][k] for this lane's kernel point k = lane%16; K(dim)=neighbor e,
        // padded 16->32 (elements 8..15 = 0). Lane half selects e base (0 or 8).
        v16bf av = __builtin_bit_cast(v16bf, z32);
        const int ebase = hi * 8;
        #pragma unroll
        for (int j = 0; j < 8; ++j) {
            const int e = ebase + j;
            const float dx = REL[e*3+0] - kx;
            const float dy = REL[e*3+1] - ky;
            const float dz = REL[e*3+2] - kz;
            const float dist = sqrtf(dx*dx + dy*dy + dz*dz);
            float w = 1.0f - dist;                            // EXTENT = 1
            w = w > 0.f ? w : 0.f;
            av[j] = __builtin_bit_cast(__bf16, f2bfbits(w));
        }

        // 4 WMMAs: fk(16x64) = w^T(16x32,pad) @ fx(32x16 per tile)
        #pragma unroll
        for (int nt = 0; nt < 4; ++nt) {
            S32 bs = {};                                      // elements 8..15 (K=16..31) are zero
            bs.lo = *(const v4u*)(FXS + (nt * 16 + lo16) * 16 + hi * 8);
            v16bf bv = __builtin_bit_cast(v16bf, bs);
            const v8f d = __builtin_amdgcn_wmma_f32_16x16x32_bf16(
                false, av, false, bv, (short)0, zc, false, false);
            unsigned short* fkrow = FK + p * FK_STRIDE_US;
            #pragma unroll
            for (int v = 0; v < 8; ++v) {
                const int k = v + hi * 8;                     // D: M = v + 8*(lane>=16)
                fkrow[k * 64 + nt * 16 + lo16] = f2bfbits(d[v]);
            }
        }
        // DS ops issue in order within a wave: next iteration's FXS writes cannot
        // pass this iteration's reads; a compiler-reordering fence is enough.
        __builtin_amdgcn_wave_barrier();
    }

    wave_lds_sync();       // FK stores -> phase C reads (same wave)

    // ---------------- Phase C: OUT(16x64) = FK(16x1024) @ W(1024x64) ----------------
    v8f acc0 = zc, acc1 = zc, acc2 = zc, acc3 = zc;
    for (int kb = 0; kb < NKB; ++kb) {
        // A fragment: row m = lane%16; K chunk [kb*32 + h*16 .. +15]
        const unsigned short* arow = FK + lo16 * FK_STRIDE_US + kb * 32 + hi * 16;
        S32 ab;
        ab.lo = *(const v4u*)(arow);
        ab.hi = *(const v4u*)(arow + 8);
        const v16bf af = __builtin_bit_cast(v16bf, ab);

        const unsigned short* wb = Wl + ((kb * 4) * 32 + lane) * 16;
        const v16bf b0 = *(const v16bf*)(wb);
        const v16bf b1 = *(const v16bf*)(wb + 512);
        const v16bf b2 = *(const v16bf*)(wb + 1024);
        const v16bf b3 = *(const v16bf*)(wb + 1536);

        acc0 = __builtin_amdgcn_wmma_f32_16x16x32_bf16(false, af, false, b0, (short)0, acc0, false, false);
        acc1 = __builtin_amdgcn_wmma_f32_16x16x32_bf16(false, af, false, b1, (short)0, acc1, false, false);
        acc2 = __builtin_amdgcn_wmma_f32_16x16x32_bf16(false, af, false, b2, (short)0, acc2, false, false);
        acc3 = __builtin_amdgcn_wmma_f32_16x16x32_bf16(false, af, false, b3, (short)0, acc3, false, false);
    }

    if (active) {
        #pragma unroll
        for (int v = 0; v < 8; ++v) {
            const int m = v + hi * 8;
            float* orow = out + (size_t)(tile * 16 + m) * COUT + lo16;
            orow[0]  = acc0[v];
            orow[16] = acc1[v];
            orow[32] = acc2[v];
            orow[48] = acc3[v];
        }
    }
}

// =====================================================================
// BN epilogue: partial sums -> finalize -> apply (+ leaky ReLU) in place
// =====================================================================
__global__ __launch_bounds__(256)
void bn_partial(const float* __restrict__ out, float* __restrict__ ws)
{
    __shared__ float red[2][256];
    const int b = blockIdx.x;             // 240 blocks * 625 rows = 150000
    const int t = threadIdx.x;
    const int c = t & 63;
    const int g = t >> 6;                 // 0..3
    const int row0 = b * 625;
    const int row1 = row0 + 625;
    float s1 = 0.f, s2 = 0.f;
    for (int r = row0 + g; r < row1; r += 4) {
        const float v = out[(size_t)r * 64 + c];
        s1 += v; s2 += v * v;
    }
    red[0][t] = s1; red[1][t] = s2;
    __syncthreads();
    if (t < 64) {
        const float a1 = red[0][t] + red[0][64+t] + red[0][128+t] + red[0][192+t];
        const float a2 = red[1][t] + red[1][64+t] + red[1][128+t] + red[1][192+t];
        ws[b * 128 + t]      = a1;
        ws[b * 128 + 64 + t] = a2;
    }
}

__global__ void bn_finalize(const float* __restrict__ part, float* __restrict__ stats)
{
    const int c = threadIdx.x;            // 64 threads
    float s1 = 0.f, s2 = 0.f;
    for (int b = 0; b < 240; ++b) {
        s1 += part[b * 128 + c];
        s2 += part[b * 128 + 64 + c];
    }
    const float inv_n = 1.0f / (float)N_PTS;
    const float mean  = s1 * inv_n;
    const float var   = s2 * inv_n - mean * mean;
    stats[c]      = mean;
    stats[64 + c] = rsqrtf(var + 1e-5f);
}

__global__ __launch_bounds__(256)
void bn_apply(float* __restrict__ out, const float* __restrict__ stats,
              const float* __restrict__ gamma, const float* __restrict__ beta)
{
    __shared__ float sm[256];
    const int t = threadIdx.x;
    if (t < 64) {
        sm[t]       = stats[t];
        sm[64 + t]  = stats[64 + t];
        sm[128 + t] = gamma[t];
        sm[192 + t] = beta[t];
    }
    __syncthreads();
    const size_t i = (size_t)blockIdx.x * 256 + t;    // grid sized exactly: N*64/256
    const int c = (int)(i & 63);
    const float v = out[i];
    const float y = (v - sm[c]) * sm[64 + c] * sm[128 + c] + sm[192 + c];
    out[i] = y > 0.f ? y : 0.1f * y;
}

// =====================================================================
extern "C" void kernel_launch(void* const* d_in, const int* in_sizes, int n_in,
                              void* d_out, int out_size, void* d_ws, size_t ws_size,
                              hipStream_t stream)
{
    const float* pos   = (const float*)d_in[0];
    const float* x     = (const float*)d_in[1];
    const int*   eidx  = (const int*)d_in[2];
    const float* kp    = (const float*)d_in[3];
    const float* Wg    = (const float*)d_in[4];
    const float* gamma = (const float*)d_in[5];
    const float* beta  = (const float*)d_in[6];
    float* out = (float*)d_out;
    float* ws  = (float*)d_ws;

    const int wgs = (NTILES + WPG - 1) / WPG;         // 2344 workgroups * 4 waves
    kpconv_main<<<wgs, 128, SMEM_BYTES, stream>>>(pos, x, eidx, kp, Wg, out);
    bn_partial<<<240, 256, 0, stream>>>(out, ws);
    bn_finalize<<<1, 64, 0, stream>>>(ws, ws + 30720);
    bn_apply<<<(N_PTS * COUT) / 256, 256, 0, stream>>>(out, ws + 30720, gamma, beta);
}
